// Model_39676907887620
// MI455X (gfx1250) — matrix operand order repaired and checked by reading, not compiled, not run on hardware
//
#include <hip/hip_runtime.h>

// ---------------------------------------------------------------------------
// MI455X / gfx1250: bf16 WMMA (v_wmma_f32_16x16x32_bf16), f32 accumulation.
// B=8, S=2048, D=128.  s = q@k^T ; a = softmax(s/scale) ; out = (a@s)@s
// ---------------------------------------------------------------------------

typedef __bf16 bf16_t;
typedef __attribute__((ext_vector_type(16))) __bf16 v16bf;
typedef __attribute__((ext_vector_type(8)))  float  v8f;
typedef __attribute__((ext_vector_type(4)))  float  f32x4;

#define B_ 8
#define S_ 2048
#define D_ 128

// ---- fragment loaders ------------------------------------------------------
// A fragment (16x32 bf16, row-major source, per ISA layout):
// lane 0-15: row M=lane,   elems 0..7 = K 0..7,   elems 8..15 = K 16..23
// lane16-31: row M=lane-16,elems 0..7 = K 8..15,  elems 8..15 = K 24..31
// caller passes p = rowBase + kBase + ((lane>>4)<<3)
__device__ __forceinline__ v16bf load_a_frag(const bf16_t* p) {
    union { f32x4 f[2]; v16bf v; } u;
    u.f[0] = *reinterpret_cast<const f32x4*>(p);        // 8 bf16: K+0..7
    u.f[1] = *reinterpret_cast<const f32x4*>(p + 16);   // 8 bf16: K+16..23
    return u.v;
}

// B fragment (32x16 bf16) from a TRANSPOSED values matrix VT (vt[n][k]):
// lane 0-15: col N=lane,   elems 0..7 = K 0..7,   elems 8..15 = K 16..23
// lane16-31: col N=lane-16,elems 0..7 = K 8..15,  elems 8..15 = K 24..31
// caller passes p = VT + n*ld + kBase + ((lane>>4)<<3)
__device__ __forceinline__ v16bf load_b_frag(const bf16_t* p) {
    union { f32x4 f[2]; v16bf v; } u;
    u.f[0] = *reinterpret_cast<const f32x4*>(p);
    u.f[1] = *reinterpret_cast<const f32x4*>(p + 16);
    return u.v;
}

#define ZERO8 {0.f,0.f,0.f,0.f,0.f,0.f,0.f,0.f}

// ---------------------------------------------------------------------------
// fp32 -> bf16 convert
// ---------------------------------------------------------------------------
__global__ __launch_bounds__(256) void f32_to_bf16(const float* __restrict__ in,
                                                   bf16_t* __restrict__ out, int n) {
    int i = blockIdx.x * 256 + threadIdx.x;
    if (i < n) out[i] = (bf16_t)in[i];
}

// ---------------------------------------------------------------------------
// GEMM1: s[b] = q[b] @ k[b]^T   (M=N=S, K=D)
// stores s row-major bf16 (for softmax) AND s^T bf16 (B-operand layout).
// block = 256 threads = 8 waves, block tile 128x128, wave tile 64x32.
// ---------------------------------------------------------------------------
__global__ __launch_bounds__(256) void gemm_qk(const bf16_t* __restrict__ Q,
                                               const bf16_t* __restrict__ Km,
                                               bf16_t* __restrict__ Srm,
                                               bf16_t* __restrict__ ST) {
    const int lane = threadIdx.x & 31;
    const int wave = threadIdx.x >> 5;
    const int wm = wave >> 2, wn = wave & 3;
    const int mW = blockIdx.x * 128 + wm * 64;
    const int nW = blockIdx.y * 128 + wn * 32;
    const size_t bQK = (size_t)blockIdx.z * S_ * D_;
    const size_t bSS = (size_t)blockIdx.z * S_ * S_;
    const bf16_t* Qb = Q  + bQK;
    const bf16_t* Kb = Km + bQK;

    const int lRow = lane & 15;
    const int aK   = (lane >> 4) << 3;   // 0 or 8
    const int bK   = (lane >> 4) << 3;   // 0 or 8

    v8f acc[4][2] = {{ZERO8,ZERO8},{ZERO8,ZERO8},{ZERO8,ZERO8},{ZERO8,ZERO8}};

#pragma unroll
    for (int k0 = 0; k0 < D_; k0 += 32) {
        v16bf bfr[2];
#pragma unroll
        for (int ni = 0; ni < 2; ++ni)
            bfr[ni] = load_b_frag(Kb + (size_t)(nW + ni*16 + lRow) * D_ + k0 + bK);
#pragma unroll
        for (int mi = 0; mi < 4; ++mi) {
            v16bf afr = load_a_frag(Qb + (size_t)(mW + mi*16 + lRow) * D_ + k0 + aK);
#pragma unroll
            for (int ni = 0; ni < 2; ++ni)
                acc[mi][ni] = __builtin_amdgcn_wmma_f32_16x16x32_bf16(
                    false, afr, false, bfr[ni], (short)0, acc[mi][ni], false, false);
        }
    }

    // C/D layout: VGPR r -> M = mOff2 + r, N = lane&15 (mOff2 = 8 for upper half-wave)
    const int mOff2 = (lane >> 4) << 3;
    bf16_t* Srmb = Srm + bSS;
    bf16_t* STb  = ST  + bSS;
#pragma unroll
    for (int mi = 0; mi < 4; ++mi) {
#pragma unroll
        for (int ni = 0; ni < 2; ++ni) {
            const int n  = nW + ni*16 + lRow;
            const int m0 = mW + mi*16 + mOff2;
            union { bf16_t h[8]; f32x4 f; } pk;
#pragma unroll
            for (int r = 0; r < 8; ++r) {
                const bf16_t v = (bf16_t)acc[mi][ni][r];
                pk.h[r] = v;
                Srmb[(size_t)(m0 + r) * S_ + n] = v;          // row-major
            }
            *reinterpret_cast<f32x4*>(STb + (size_t)n * S_ + m0) = pk.f;  // transposed, 16B store
        }
    }
}

// ---------------------------------------------------------------------------
// softmax over rows of s (scale read from device scalar): a = softmax(s/scale)
// one block (256 threads) per row of 2048.
// ---------------------------------------------------------------------------
__global__ __launch_bounds__(256) void softmax_rows(const bf16_t* __restrict__ Srm,
                                                    bf16_t* __restrict__ Aout,
                                                    const int* __restrict__ scaleP) {
    __shared__ float red[256];
    const int tid = threadIdx.x;
    const size_t row = blockIdx.x;
    const bf16_t* src = Srm + row * S_;
    bf16_t* dst = Aout + row * S_;
    const float invS = 1.0f / (float)(*scaleP);

    float x[8];
#pragma unroll
    for (int i = 0; i < 8; ++i) x[i] = (float)src[tid + i * 256];

    float m = x[0];
#pragma unroll
    for (int i = 1; i < 8; ++i) m = fmaxf(m, x[i]);
    red[tid] = m; __syncthreads();
    for (int off = 128; off > 0; off >>= 1) {
        if (tid < off) red[tid] = fmaxf(red[tid], red[tid + off]);
        __syncthreads();
    }
    const float rmax = red[0]; __syncthreads();

    float e[8], s = 0.f;
#pragma unroll
    for (int i = 0; i < 8; ++i) { e[i] = __expf((x[i] - rmax) * invS); s += e[i]; }
    red[tid] = s; __syncthreads();
    for (int off = 128; off > 0; off >>= 1) {
        if (tid < off) red[tid] += red[tid + off];
        __syncthreads();
    }
    const float inv = 1.0f / red[0];
#pragma unroll
    for (int i = 0; i < 8; ++i) dst[tid + i * 256] = (bf16_t)(e[i] * inv);
}

// ---------------------------------------------------------------------------
// GEMM: C[b] = A[b] @ V[b]  with V given transposed (VT[n][k] = V[k][n]).
// M=N=K=S.  OUTF32=0: store bf16 row-major; OUTF32=1: store f32 row-major.
// ---------------------------------------------------------------------------
template <int OUTF32>
__global__ __launch_bounds__(256) void gemm_sv(const bf16_t* __restrict__ Amat,
                                               const bf16_t* __restrict__ VT,
                                               void* __restrict__ Cout) {
    const int lane = threadIdx.x & 31;
    const int wave = threadIdx.x >> 5;
    const int wm = wave >> 2, wn = wave & 3;
    const int mW = blockIdx.x * 128 + wm * 64;
    const int nW = blockIdx.y * 128 + wn * 32;
    const size_t bSS = (size_t)blockIdx.z * S_ * S_;
    const bf16_t* Ab = Amat + bSS;
    const bf16_t* Vb = VT   + bSS;

    const int lRow = lane & 15;
    const int aK   = (lane >> 4) << 3;
    const int bK   = (lane >> 4) << 3;

    v8f acc[4][2] = {{ZERO8,ZERO8},{ZERO8,ZERO8},{ZERO8,ZERO8},{ZERO8,ZERO8}};

#pragma unroll 2
    for (int k0 = 0; k0 < S_; k0 += 32) {
        // L2-resident working set (8 MB/batch << 192 MB L2): prefetch next chunk
        __builtin_prefetch(Ab + (size_t)(mW + lRow) * S_ + k0 + 64, 0, 3);
        v16bf bfr[2];
#pragma unroll
        for (int ni = 0; ni < 2; ++ni)
            bfr[ni] = load_b_frag(Vb + (size_t)(nW + ni*16 + lRow) * S_ + k0 + bK);
#pragma unroll
        for (int mi = 0; mi < 4; ++mi) {
            v16bf afr = load_a_frag(Ab + (size_t)(mW + mi*16 + lRow) * S_ + k0 + aK);
#pragma unroll
            for (int ni = 0; ni < 2; ++ni)
                acc[mi][ni] = __builtin_amdgcn_wmma_f32_16x16x32_bf16(
                    false, afr, false, bfr[ni], (short)0, acc[mi][ni], false, false);
        }
    }

    const int mOff2 = (lane >> 4) << 3;
#pragma unroll
    for (int mi = 0; mi < 4; ++mi) {
#pragma unroll
        for (int ni = 0; ni < 2; ++ni) {
            const int n  = nW + ni*16 + lRow;
            const int m0 = mW + mi*16 + mOff2;
            if (OUTF32) {
                float* C = (float*)Cout + bSS;
#pragma unroll
                for (int r = 0; r < 8; ++r)
                    C[(size_t)(m0 + r) * S_ + n] = acc[mi][ni][r];
            } else {
                bf16_t* C = (bf16_t*)Cout + bSS;
#pragma unroll
                for (int r = 0; r < 8; ++r)
                    C[(size_t)(m0 + r) * S_ + n] = (bf16_t)acc[mi][ni][r];
            }
        }
    }
}

// ---------------------------------------------------------------------------
// launch
// ---------------------------------------------------------------------------
extern "C" void kernel_launch(void* const* d_in, const int* in_sizes, int n_in,
                              void* d_out, int out_size, void* d_ws, size_t ws_size,
                              hipStream_t stream) {
    const float* q1 = (const float*)d_in[0];
    const float* k1 = (const float*)d_in[1];
    const int* inv_scale = (const int*)d_in[3];   // inver_scale_factor (device scalar)
    float* out = (float*)d_out;

    const size_t nQK = (size_t)B_ * S_ * D_;      // 2,097,152
    const size_t nSS = (size_t)B_ * S_ * S_;      // 33,554,432

    bf16_t* qb   = (bf16_t*)d_ws;                 // ws layout (bf16): ~264 MiB total
    bf16_t* kb   = qb   + nQK;
    bf16_t* s_rm = kb   + nQK;                    // s row-major
    bf16_t* sT   = s_rm + nSS;                    // s transposed (B-operand layout)
    bf16_t* a    = sT   + nSS;                    // softmax(s/scale)
    bf16_t* t    = a    + nSS;                    // a @ s

    f32_to_bf16<<<(unsigned)((nQK + 255) / 256), 256, 0, stream>>>(q1, qb, (int)nQK);
    f32_to_bf16<<<(unsigned)((nQK + 255) / 256), 256, 0, stream>>>(k1, kb, (int)nQK);

    dim3 g(S_ / 128, S_ / 128, B_);               // (16,16,8)
    gemm_qk<<<g, 256, 0, stream>>>(qb, kb, s_rm, sT);

    softmax_rows<<<B_ * S_, 256, 0, stream>>>(s_rm, a, inv_scale);

    gemm_sv<0><<<g, 256, 0, stream>>>(a, sT, (void*)t);     // t = a @ s   (bf16)
    gemm_sv<1><<<g, 256, 0, stream>>>(t, sT, (void*)out);   // out = t @ s (f32)
}
